// RetinaNet_47339129536784
// MI455X (gfx1250) — 8-bit matrix operand order repaired and checked by reading, compiled, not run on hardware
//
#include <hip/hip_runtime.h>
#include <hip/hip_bf16.h>

typedef __attribute__((ext_vector_type(8))) int v8i;

#define B_IMG 32
#define TOPK 1000
#define KPAD 1024
#define CANDMAX 2048
#define NCLS 80
#define MAXOUT 100

__device__ __forceinline__ unsigned oneh4(int d) {
    return (d >= 0 && d < 4) ? (1u << (8 * d)) : 0u;
}

__global__ __launch_bounds__(1024) void topk_kernel(
    const float* __restrict__ boxes,
    const int* __restrict__ classes,
    const float* __restrict__ conf,
    int N,
    float* __restrict__ wsScore,
    int* __restrict__ wsCls,
    float* __restrict__ wsBox)
{
    const int b = blockIdx.x;
    const int tid = threadIdx.x;
    const int NT = blockDim.x;

    __shared__ unsigned hist[256];
    __shared__ unsigned sPrefix, sMask, sRemK, sCount;
    __shared__ unsigned long long cand[CANDMAX];

    const float* sc = conf + (size_t)b * N;

    if (tid == 0) { sPrefix = 0u; sMask = 0u; sRemK = TOPK; }
    __syncthreads();

    for (int shift = 24; shift >= 0; shift -= 8) {
        if (tid < 256) hist[tid] = 0u;
        __syncthreads();
        unsigned pfx = sPrefix, msk = sMask;
        for (int e = tid; e < N; e += NT) {
            unsigned k = __float_as_uint(sc[e]);
            if ((k & msk) == pfx) atomicAdd(&hist[(k >> shift) & 255u], 1u);
        }
        __syncthreads();
        if (tid == 0) {
            unsigned need = sRemK, cum = 0u;
            int d = 255;
            for (; d > 0; --d) { cum += hist[d]; if (cum >= need) break; }
            if (cum < need) cum += hist[0];
            sRemK = need - (cum - hist[d]);
            sPrefix |= ((unsigned)d) << shift;
            sMask |= 0xFFu << shift;
        }
        __syncthreads();
    }
    if (tid == 0) sCount = 0u;
    __syncthreads();
    const unsigned T = sPrefix;

    for (int e = tid; e < N; e += NT) {
        unsigned k = __float_as_uint(sc[e]);
        if (k >= T) {
            unsigned p = atomicAdd(&sCount, 1u);
            if (p < CANDMAX)
                cand[p] = ((unsigned long long)k << 32) | (unsigned)(~(unsigned)e);
        }
    }
    __syncthreads();
    unsigned cnt = sCount; if (cnt > CANDMAX) cnt = CANDMAX;
    for (int p = (int)cnt + tid; p < CANDMAX; p += NT) cand[p] = 0ull;
    __syncthreads();

    for (int k = 2; k <= CANDMAX; k <<= 1) {
        for (int j = k >> 1; j > 0; j >>= 1) {
            for (int i = tid; i < CANDMAX; i += NT) {
                int ixj = i ^ j;
                if (ixj > i) {
                    bool desc = ((i & k) == 0);
                    unsigned long long a = cand[i], c = cand[ixj];
                    if (desc ? (a < c) : (a > c)) { cand[i] = c; cand[ixj] = a; }
                }
            }
            __syncthreads();
        }
    }

    for (int t = tid; t < KPAD; t += NT) {
        if (t < TOPK) {
            unsigned long long e = cand[t];
            unsigned key = (unsigned)(e >> 32);
            int idx = (int)(~(unsigned)e);
            wsScore[b * KPAD + t] = __uint_as_float(key);
            wsCls[b * KPAD + t] = classes[(size_t)b * N + idx];
            const float* bp = boxes + ((size_t)b * N + idx) * 4;
            float4 bb; bb.x = bp[0]; bb.y = bp[1]; bb.z = bp[2]; bb.w = bp[3];
            ((float4*)wsBox)[b * KPAD + t] = bb;
        } else {
            wsScore[b * KPAD + t] = -1.0f;
            wsCls[b * KPAD + t] = 1 << 20;
            float4 z; z.x = z.y = z.z = z.w = 0.0f;
            ((float4*)wsBox)[b * KPAD + t] = z;
        }
    }
}

#define NMS_NT 512
#define SM_BOX    0
#define SM_SCORE  (SM_BOX + KPAD * 16)
#define SM_AREA   (SM_SCORE + KPAD * 4)
#define SM_CLS    (SM_AREA + KPAD * 4)
#define SM_SUPP   (SM_CLS + KPAD * 4)
#define SM_KEEP   (SM_SUPP + KPAD * 32 * 4)
#define SM_VALID  (SM_KEEP + 32 * 4)
#define SM_CMAX   (SM_VALID + KPAD * 4)
#define SM_TOTAL  (SM_CMAX + NCLS * 4)

__global__ __launch_bounds__(NMS_NT) void nms_kernel(
    const float* __restrict__ wsScore,
    const int* __restrict__ wsCls,
    const float* __restrict__ wsBox,
    float* __restrict__ dout)
{
    extern __shared__ char smem[];
    float4* sBox      = (float4*)(smem + SM_BOX);
    float* sScore     = (float*)(smem + SM_SCORE);
    float* sArea      = (float*)(smem + SM_AREA);
    int* sCls         = (int*)(smem + SM_CLS);
    unsigned* suppRow = (unsigned*)(smem + SM_SUPP);
    unsigned* keepW   = (unsigned*)(smem + SM_KEEP);
    int* validA       = (int*)(smem + SM_VALID);
    float* classMax   = (float*)(smem + SM_CMAX);

    const int b = blockIdx.x;
    const int tid = threadIdx.x;
    const int lane = tid & 31;
    const int wave = tid >> 5;
    const int WPB = NMS_NT / 32;

    for (int t = tid; t < KPAD; t += NMS_NT) {
        float4 bb = ((const float4*)wsBox)[b * KPAD + t];
        sBox[t] = bb;
        sScore[t] = wsScore[b * KPAD + t];
        sCls[t] = wsCls[b * KPAD + t];
        sArea[t] = (bb.z - bb.x) * (bb.w - bb.y);
    }
    __syncthreads();

    const int m = lane & 15;
    const int half = lane >> 4;

    auto buildB = [&](int J2, v8i& B0, v8i& B1) {
        const int cb0 = sCls[(J2 * 2) * 16 + m];
        const int cb1 = sCls[(J2 * 2 + 1) * 16 + m];
        const int h0 = cb0 >> 3, l0 = (cb0 & 7) + 16;
        const int h1 = cb1 >> 3, l1 = (cb1 & 7) + 16;
#pragma unroll
        for (int v = 0; v < 8; ++v) {
            int kb = ((v >> 1) << 4) + ((v & 1) << 2) + (half << 3);
            B0[v] = (int)(oneh4(h0 - kb) | oneh4(l0 - kb));
            B1[v] = (int)(oneh4(h1 - kb) | oneh4(l1 - kb));
        }
    };

    auto consume = [&](int I, int J2, const v8i& C0, const v8i& C1) {
        const int j0 = (J2 * 2) * 16 + m;
        const float4 bj0 = sBox[j0];
        const float4 bj1 = sBox[j0 + 16];
        const float aj0 = sArea[j0];
        const float aj1 = sArea[j0 + 16];
#pragma unroll
        for (int r = 0; r < 8; ++r) {
            int i = I * 16 + r + half * 8;
            float4 bi = sBox[i];
            float ai = sArea[i];
            float ix0 = fminf(bi.z, bj0.z) - fmaxf(bi.x, bj0.x);
            float iy0 = fminf(bi.w, bj0.w) - fmaxf(bi.y, bj0.y);
            float in0 = fmaxf(ix0, 0.0f) * fmaxf(iy0, 0.0f);
            float ix1 = fminf(bi.z, bj1.z) - fmaxf(bi.x, bj1.x);
            float iy1 = fminf(bi.w, bj1.w) - fmaxf(bi.y, bj1.y);
            float in1 = fmaxf(ix1, 0.0f) * fmaxf(iy1, 0.0f);
            bool bit0 = (3.0f * in0 > ai + aj0) && (C0[r] == 2);
            bool bit1 = (3.0f * in1 > ai + aj1) && (C1[r] == 2);
#if __has_builtin(__builtin_amdgcn_ballot_w32)
            unsigned bal0 = __builtin_amdgcn_ballot_w32(bit0);
            unsigned bal1 = __builtin_amdgcn_ballot_w32(bit1);
#else
            unsigned bal0 = (unsigned)__ballot(bit0);
            unsigned bal1 = (unsigned)__ballot(bit1);
#endif
            if (lane == 0) {
                suppRow[(I * 16 + r) * 32 + J2] = (bal0 & 0xFFFFu) | (bal1 << 16);
                suppRow[(I * 16 + r + 8) * 32 + J2] = (bal0 >> 16) | (bal1 & 0xFFFF0000u);
            }
        }
    };

    for (int I = wave; I < 64; I += WPB) {
        const int ca = sCls[I * 16 + m];
        const int hiA = ca >> 3;
        const int loA = (ca & 7) + 16;
        v8i A;
#pragma unroll
        for (int v = 0; v < 8; ++v) {
            int kb = ((v >> 1) << 4) + ((v & 1) << 2) + (half << 3);
            A[v] = (int)(oneh4(hiA - kb) | oneh4(loA - kb));
        }

        const v8i Cz = {0, 0, 0, 0, 0, 0, 0, 0};
        v8i B0, B1;
        buildB(0, B0, B1);
        v8i C0 = __builtin_amdgcn_wmma_i32_16x16x64_iu8(false, A, false, B0, Cz, false, false);
        v8i C1 = __builtin_amdgcn_wmma_i32_16x16x64_iu8(false, A, false, B1, Cz, false, false);
#pragma unroll 2
        for (int J2 = 0; J2 < 31; ++J2) {
            v8i Bn0, Bn1;
            buildB(J2 + 1, Bn0, Bn1);
            v8i N0 = __builtin_amdgcn_wmma_i32_16x16x64_iu8(false, A, false, Bn0, Cz, false, false);
            v8i N1 = __builtin_amdgcn_wmma_i32_16x16x64_iu8(false, A, false, Bn1, Cz, false, false);
            consume(I, J2, C0, C1);
            C0 = N0; C1 = N1;
        }
        consume(I, 31, C0, C1);
    }
    __syncthreads();

    if (wave == 0) {
        unsigned supp = 0u, keep = 0u;
        for (int i = 0; i < TOPK; ++i) {
            unsigned w = (unsigned)__builtin_amdgcn_readlane((int)supp, i >> 5);
            bool active = ((w >> (i & 31)) & 1u) == 0u;
            if (active) {
                if (lane == (i >> 5)) keep |= 1u << (i & 31);
                supp |= suppRow[i * 32 + lane];
            }
        }
        keepW[lane] = keep;
    }
    __syncthreads();

    const float thr1 = 0.1f * sScore[0];
    for (int j = tid; j < KPAD; j += NMS_NT) {
        int v = (j < TOPK) && ((keepW[j >> 5] >> (j & 31)) & 1u) && (sScore[j] >= thr1);
        validA[j] = v;
    }
    __syncthreads();

    if (tid == 0) {
        for (int c = 0; c < NCLS; ++c) classMax[c] = -1.0f;
        float* ob = dout + (size_t)b * (MAXOUT * 4);
        float* oc = dout + (size_t)B_IMG * MAXOUT * 4 + (size_t)b * MAXOUT;
        float* os = dout + (size_t)B_IMG * MAXOUT * 5 + (size_t)b * MAXOUT;
        int cnt = 0;
        for (int j = 0; j < TOPK; ++j) {
            if (!validA[j]) continue;
            int c = sCls[j];
            float s = sScore[j];
            bool inRange = (c >= 0 && c < NCLS);
            bool del = inRange && (s < 0.05f * classMax[c]);
            if (!del && cnt < MAXOUT) {
                float4 bb = sBox[j];
                ob[cnt * 4 + 0] = bb.x; ob[cnt * 4 + 1] = bb.y;
                ob[cnt * 4 + 2] = bb.z; ob[cnt * 4 + 3] = bb.w;
                oc[cnt] = (float)c;
                os[cnt] = s;
                ++cnt;
            }
            if (inRange && s > classMax[c]) classMax[c] = s;
        }
        for (; cnt < MAXOUT; ++cnt) {
            ob[cnt * 4 + 0] = 0.0f; ob[cnt * 4 + 1] = 0.0f;
            ob[cnt * 4 + 2] = 0.0f; ob[cnt * 4 + 3] = 0.0f;
            oc[cnt] = -1.0f;
            os[cnt] = 0.0f;
        }
    }
}

extern "C" void kernel_launch(void* const* d_in, const int* in_sizes, int n_in,
                              void* d_out, int out_size, void* d_ws, size_t ws_size,
                              hipStream_t stream) {
    const float* boxes = (const float*)d_in[0];
    const int* classes = (const int*)d_in[1];
    const float* conf = (const float*)d_in[2];
    const int N = in_sizes[1] / B_IMG;

    float* wsScore = (float*)d_ws;
    int* wsCls = (int*)((char*)d_ws + (size_t)B_IMG * KPAD * 4);
    float* wsBox = (float*)((char*)d_ws + (size_t)B_IMG * KPAD * 8);

    topk_kernel<<<B_IMG, 1024, 0, stream>>>(boxes, classes, conf, N, wsScore, wsCls, wsBox);
    nms_kernel<<<B_IMG, NMS_NT, SM_TOTAL, stream>>>(wsScore, wsCls, wsBox, (float*)d_out);
}
